// GatedConvNeXt3DBlock_19628000543413
// MI455X (gfx1250) — compile-verified, matrix operand order checked by reading, not run on hardware
//
#include <hip/hip_runtime.h>

// ---------------------------------------------------------------------------
// GatedConvNeXt3DBlock for MI455X (gfx1250, wave32, WMMA bf16)
// B=2, T=16, H=56, W=56, C=128, HIDDEN=512, kernel 5x7x7 circular depthwise
// ---------------------------------------------------------------------------

#define Bdim 2
#define Tdim 16
#define Hdim 56
#define Wdim 56
#define Cdim 128
#define HID 512
#define NROWS (Bdim*Tdim*Hdim*Wdim)        // 100352
#define HWdim (Hdim*Wdim)                   // 3136

typedef __attribute__((ext_vector_type(16))) __bf16 v16bf;
typedef __attribute__((ext_vector_type(8)))  __bf16 v8bf;
typedef __attribute__((ext_vector_type(8)))  float  v8f;

// ---- bf16 helpers (round-to-nearest-even) ----
__device__ __forceinline__ unsigned short f2bf(float f) {
    unsigned u = __float_as_uint(f);
    u += 0x7FFFu + ((u >> 16) & 1u);
    return (unsigned short)(u >> 16);
}
__device__ __forceinline__ float bf2f(unsigned short s) {
    return __uint_as_float(((unsigned)s) << 16);
}

// ---- WMMA fragment loader: row-major bf16, 16-bit A/B layout ----
// lane 0-15:  row = lane,    elems 0..7 -> K0..K7,   8..15 -> K16..23
// lane 16-31: row = lane-16, elems 0..7 -> K8..K15,  8..15 -> K24..31
// caller passes p = base + row*ld + ((lane&16)?8:0) + ktile
__device__ __forceinline__ v16bf load_frag(const unsigned short* p) {
    v8bf lo = *(const v8bf*)(p);
    v8bf hi = *(const v8bf*)(p + 16);
    return __builtin_shufflevector(lo, hi, 0,1,2,3,4,5,6,7,8,9,10,11,12,13,14,15);
}

__device__ __forceinline__ float softplusf(float x) {
    return fmaxf(x, 0.f) + log1pf(expf(-fabsf(x)));
}
__device__ __forceinline__ float logaddexpf_(float p, float q) {
    float m = fmaxf(p, q);
    return m + log1pf(expf(-fabsf(p - q)));
}
__device__ __forceinline__ float geluf(float x) {
    float t = tanhf(0.7978845608f * (x + 0.044715f * x * x * x));
    return 0.5f * x * (1.f + t);
}

// ---------------------------------------------------------------------------
// 0) fp32 -> bf16 convert (activations)
// ---------------------------------------------------------------------------
__global__ __launch_bounds__(256) void k_convert_bf16(const float* __restrict__ in,
                                                      unsigned short* __restrict__ out,
                                                      int n) {
    int i = blockIdx.x * 256 + threadIdx.x;
    if (i < n) out[i] = f2bf(in[i]);
}

// 0b) transpose fp32 (rows x cols) -> bf16 (cols x rows)
__global__ __launch_bounds__(256) void k_transpose_bf(const float* __restrict__ in,
                                                      unsigned short* __restrict__ out,
                                                      int rows, int cols) {
    int i = blockIdx.x * 256 + threadIdx.x;
    if (i >= rows * cols) return;
    int r = i / cols, c = i - r * cols;
    out[(size_t)c * rows + r] = f2bf(in[i]);
}

// ---------------------------------------------------------------------------
// 1) Dual projection GEMM: gate_proj = x@Wg+bg ; hidden_proj = x@Wh+bh
//    block = 16 rows x 128 cols, 8 waves, one 16x16 tile per wave, K=128
// ---------------------------------------------------------------------------
__global__ __launch_bounds__(256) void k_dual_gemm_proj(
    const unsigned short* __restrict__ A,    // NROWS x 128 bf16
    const unsigned short* __restrict__ BgT,  // 128 x 128 bf16 (Wg^T)
    const unsigned short* __restrict__ BhT,
    const float* __restrict__ bg, const float* __restrict__ bh,
    unsigned short* __restrict__ Og, unsigned short* __restrict__ Oh)
{
    int m0   = blockIdx.x * 16;
    int wave = threadIdx.x >> 5;
    int lane = threadIdx.x & 31;
    int n0   = wave * 16;
    int l15  = lane & 15;
    int kh   = (lane & 16) ? 8 : 0;

    const unsigned short* arow = A   + (size_t)(m0 + l15) * Cdim + kh;
    const unsigned short* bgr  = BgT + (size_t)(n0 + l15) * Cdim + kh;
    const unsigned short* bhr  = BhT + (size_t)(n0 + l15) * Cdim + kh;

    v8f accG = {}; v8f accH = {};
#pragma unroll
    for (int kk = 0; kk < 4; ++kk) {
        v16bf a   = load_frag(arow + kk * 32);
        v16bf bgf = load_frag(bgr + kk * 32);
        v16bf bhf = load_frag(bhr + kk * 32);
        accG = __builtin_amdgcn_wmma_f32_16x16x32_bf16(false, a, false, bgf, (short)0, accG, false, false);
        accH = __builtin_amdgcn_wmma_f32_16x16x32_bf16(false, a, false, bhf, (short)0, accH, false, false);
    }
    int n = n0 + l15;
    int mb = m0 + ((lane & 16) ? 8 : 0);
    float vg = bg[n], vh = bh[n];
#pragma unroll
    for (int r = 0; r < 8; ++r) {
        size_t idx = (size_t)(mb + r) * Cdim + n;
        Og[idx] = f2bf(accG[r] + vg);
        Oh[idx] = f2bf(accH[r] + vh);
    }
}

// ---------------------------------------------------------------------------
// 2) Fused depthwise circular 3D conv 5x7x7 for BOTH gate & hidden branches
//    (spatial-domain equivalent of the FFT path). Tap geometry is shared, so
//    index math is computed once and amortized over 2 loads + 2 FMAs.
//    y[t,h,w,c] = sum ker[c,dt,dh,dw] * x[(t-dt+2)%T,(h-dh+3)%H,(w-dw+3)%W,c]
// ---------------------------------------------------------------------------
__global__ __launch_bounds__(256) void k_dwconv3d_dual(
    const unsigned short* __restrict__ gin,  // gate_proj (B,T,H,W,C) bf16
    const unsigned short* __restrict__ hin,  // hidden_proj
    const float* __restrict__ gker,          // (C,5,7,7)
    const float* __restrict__ hker,
    float* __restrict__ gout,                // f32
    float* __restrict__ hout)
{
    int g = blockIdx.x * 256 + threadIdx.x;   // < NROWS*Cdim (fits int)
    int c = g & (Cdim - 1);
    int s = g >> 7;
    int w = s % Wdim; s /= Wdim;
    int h = s % Hdim; s /= Hdim;
    int t = s % Tdim;
    int b = s / Tdim;

    int tt[5], hh[7], ww[7];
#pragma unroll
    for (int dt = 0; dt < 5; ++dt) { int v = t - dt + 2; if (v < 0) v += Tdim; if (v >= Tdim) v -= Tdim; tt[dt] = v; }
#pragma unroll
    for (int d = 0; d < 7; ++d) {
        int vh2 = h - d + 3; if (vh2 < 0) vh2 += Hdim; if (vh2 >= Hdim) vh2 -= Hdim; hh[d] = vh2;
        int vw2 = w - d + 3; if (vw2 < 0) vw2 += Wdim; if (vw2 >= Wdim) vw2 -= Wdim; ww[d] = vw2;
    }
    const float* gkc = gker + c * 245;
    const float* hkc = hker + c * 245;
    float accG = 0.f, accH = 0.f;
#pragma unroll
    for (int dt = 0; dt < 5; ++dt) {
        int obT = (b * Tdim + tt[dt]) * HWdim;
#pragma unroll
        for (int dh = 0; dh < 7; ++dh) {
            int ob = obT + hh[dh] * Wdim;
            const float* gkr = gkc + (dt * 7 + dh) * 7;
            const float* hkr = hkc + (dt * 7 + dh) * 7;
#pragma unroll
            for (int dw = 0; dw < 7; ++dw) {
                size_t idx = (size_t)(ob + ww[dw]) * Cdim + c;
                accG += gkr[dw] * bf2f(gin[idx]);
                accH += hkr[dw] * bf2f(hin[idx]);
            }
        }
    }
    gout[g] = accG;
    hout[g] = accH;
}

// ---------------------------------------------------------------------------
// 3) log-space temporal scan (Heinsen), T=16 sequential per (b,h,w,c)
//    a = -softplus(g); v = -softplus(-g) + log(h^2+1e-6)
//    logh_t = logaddexp(a_t + logh_{t-1}, v_t);  emit exp(logh) as bf16
// ---------------------------------------------------------------------------
__global__ __launch_bounds__(256) void k_scan(
    const float* __restrict__ gs, const float* __restrict__ hs,
    unsigned short* __restrict__ expU)
{
    int g = blockIdx.x * 256 + threadIdx.x;   // < B*H*W*C = 802816
    int c  = g & (Cdim - 1);
    int s  = g >> 7;
    int hw = s % HWdim;
    int b  = s / HWdim;
    int base = ((b * Tdim) * HWdim + hw) * Cdim + c;
    const int strideT = HWdim * Cdim;
    float logh = 0.f;
    for (int t = 0; t < Tdim; ++t) {
        int idx = base + t * strideT;
        float gv = gs[idx];
        float hv = hs[idx];
        float a  = -softplusf(gv);
        float bv = -softplusf(-gv) + logf(hv * hv + 1e-6f);
        logh = (t == 0) ? bv : logaddexpf_(a + logh, bv);
        expU[idx] = f2bf(expf(logh));
    }
}

// ---------------------------------------------------------------------------
// 4) ssm_out = expU@Wo + bo, fused LayerNorm over C -> xn (bf16)
// ---------------------------------------------------------------------------
__global__ __launch_bounds__(256) void k_gemm_wo_ln(
    const unsigned short* __restrict__ A,    // expU
    const unsigned short* __restrict__ BT,   // Wo^T
    const float* __restrict__ bo,
    const float* __restrict__ ln_scale, const float* __restrict__ ln_bias,
    unsigned short* __restrict__ Oxn)
{
    __shared__ float sm[16 * Cdim];
    __shared__ float s_mean[16], s_rstd[16];

    int m0   = blockIdx.x * 16;
    int wave = threadIdx.x >> 5;
    int lane = threadIdx.x & 31;
    int n0   = wave * 16;
    int l15  = lane & 15;
    int kh   = (lane & 16) ? 8 : 0;

    const unsigned short* arow = A  + (size_t)(m0 + l15) * Cdim + kh;
    const unsigned short* brow = BT + (size_t)(n0 + l15) * Cdim + kh;

    v8f acc = {};
#pragma unroll
    for (int kk = 0; kk < 4; ++kk) {
        v16bf a = load_frag(arow + kk * 32);
        v16bf b = load_frag(brow + kk * 32);
        acc = __builtin_amdgcn_wmma_f32_16x16x32_bf16(false, a, false, b, (short)0, acc, false, false);
    }
    int n = n0 + l15;
    int ml = (lane & 16) ? 8 : 0;
    float vb = bo[n];
#pragma unroll
    for (int r = 0; r < 8; ++r) sm[(ml + r) * Cdim + n] = acc[r] + vb;
    __syncthreads();

    if (threadIdx.x < 16) {
        const float* row = sm + threadIdx.x * Cdim;
        float s1 = 0.f, s2 = 0.f;
        for (int j = 0; j < Cdim; ++j) { float v = row[j]; s1 += v; s2 += v * v; }
        float mean = s1 * (1.f / Cdim);
        float var  = s2 * (1.f / Cdim) - mean * mean;
        s_mean[threadIdx.x] = mean;
        s_rstd[threadIdx.x] = rsqrtf(var + 1e-6f);
    }
    __syncthreads();

#pragma unroll
    for (int j = 0; j < 8; ++j) {
        int idx = threadIdx.x * 8 + j;         // 0..2047 == 16*128
        int row = idx >> 7, col = idx & (Cdim - 1);
        float v  = sm[idx];
        float xn = (v - s_mean[row]) * s_rstd[row] * ln_scale[col] + ln_bias[col];
        Oxn[(size_t)(m0 + row) * Cdim + col] = f2bf(xn);
    }
}

// ---------------------------------------------------------------------------
// 5) MLP: y = gelu(xn@W1+b1)@W2 + b2 ; out = y*gamma + residual(x)
//    stage1: 16x512 hidden in LDS (bf16), stage2: 16x128 output, both WMMA
// ---------------------------------------------------------------------------
__global__ __launch_bounds__(256) void k_mlp(
    const unsigned short* __restrict__ Xn,   // NROWS x 128 bf16
    const unsigned short* __restrict__ W1T,  // 512 x 128 bf16
    const unsigned short* __restrict__ W2T,  // 128 x 512 bf16
    const float* __restrict__ b1, const float* __restrict__ b2,
    const float* __restrict__ gamma, const float* __restrict__ xres,
    float* __restrict__ out)
{
    __shared__ __align__(16) unsigned short hid[16 * HID];

    int m0   = blockIdx.x * 16;
    int wave = threadIdx.x >> 5;
    int lane = threadIdx.x & 31;
    int l15  = lane & 15;
    int kh   = (lane & 16) ? 8 : 0;
    int ml   = (lane & 16) ? 8 : 0;

    const unsigned short* arow = Xn + (size_t)(m0 + l15) * Cdim + kh;

    // stage 1: hidden = gelu(xn @ W1 + b1)  (each wave: 4 column tiles)
#pragma unroll
    for (int ct = 0; ct < 4; ++ct) {
        int n0 = (wave * 4 + ct) * 16;
        const unsigned short* brow = W1T + (size_t)(n0 + l15) * Cdim + kh;
        v8f acc = {};
#pragma unroll
        for (int kk = 0; kk < 4; ++kk) {
            v16bf a = load_frag(arow + kk * 32);
            v16bf b = load_frag(brow + kk * 32);
            acc = __builtin_amdgcn_wmma_f32_16x16x32_bf16(false, a, false, b, (short)0, acc, false, false);
        }
        int n = n0 + l15;
        float bb = b1[n];
#pragma unroll
        for (int r = 0; r < 8; ++r)
            hid[(ml + r) * HID + n] = f2bf(geluf(acc[r] + bb));
    }
    __syncthreads();

    // stage 2: out tile = hidden @ W2, K = 512
    int n0 = wave * 16;
    const unsigned short* arow2 = hid + (size_t)l15 * HID + kh;
    const unsigned short* brow2 = W2T + (size_t)(n0 + l15) * HID + kh;
    v8f acc = {};
#pragma unroll
    for (int kk = 0; kk < 16; ++kk) {
        v16bf a = load_frag(arow2 + kk * 32);
        v16bf b = load_frag(brow2 + kk * 32);
        acc = __builtin_amdgcn_wmma_f32_16x16x32_bf16(false, a, false, b, (short)0, acc, false, false);
    }
    int n = n0 + l15;
    float bb = b2[n], gm = gamma[n];
#pragma unroll
    for (int r = 0; r < 8; ++r) {
        size_t idx = (size_t)(m0 + ml + r) * Cdim + n;
        out[idx] = (acc[r] + bb) * gm + xres[idx];
    }
}

// ---------------------------------------------------------------------------
// host launcher
// ---------------------------------------------------------------------------
extern "C" void kernel_launch(void* const* d_in, const int* in_sizes, int n_in,
                              void* d_out, int out_size, void* d_ws, size_t ws_size,
                              hipStream_t stream) {
    const float* x      = (const float*)d_in[0];
    const float* Wg     = (const float*)d_in[1];
    const float* bg     = (const float*)d_in[2];
    const float* Wh     = (const float*)d_in[3];
    const float* bh     = (const float*)d_in[4];
    const float* gker   = (const float*)d_in[5];
    const float* hker   = (const float*)d_in[6];
    const float* Wo     = (const float*)d_in[7];
    const float* bo     = (const float*)d_in[8];
    const float* lns    = (const float*)d_in[9];
    const float* lnb    = (const float*)d_in[10];
    const float* W1     = (const float*)d_in[11];
    const float* b1     = (const float*)d_in[12];
    const float* W2     = (const float*)d_in[13];
    const float* b2     = (const float*)d_in[14];
    const float* gamma  = (const float*)d_in[15];
    float* out = (float*)d_out;

    const size_t NC  = (size_t)NROWS * Cdim;
    const size_t NC2 = NC * 2;   // bf16 bytes
    const size_t NC4 = NC * 4;   // f32 bytes

    char* ws = (char*)d_ws;
    unsigned short* xbf   = (unsigned short*)(ws);                 // also expU later
    unsigned short* gproj = (unsigned short*)(ws + NC2);           // also xn later
    unsigned short* hproj = (unsigned short*)(ws + 2 * NC2);
    float*          gspat = (float*)        (ws + 3 * NC2);
    float*          hspat = (float*)        (ws + 3 * NC2 + NC4);
    char* wbase = ws + 3 * NC2 + 2 * NC4;
    unsigned short* WgT = (unsigned short*)(wbase);                 // 128x128
    unsigned short* WhT = (unsigned short*)(wbase + 32768);
    unsigned short* WoT = (unsigned short*)(wbase + 65536);
    unsigned short* W1T = (unsigned short*)(wbase + 98304);         // 512x128
    unsigned short* W2T = (unsigned short*)(wbase + 98304 + 131072);// 128x512
    unsigned short* expU = xbf;     // reuse after GEMM1 consumed xbf
    unsigned short* xn   = gproj;   // reuse after conv consumed gproj

    const int elemBlocks = (int)((NC + 255) / 256);                 // 50176
    const int rowBlocks  = NROWS / 16;                              // 6272

    // 0) converts / weight transposes
    k_convert_bf16<<<elemBlocks, 256, 0, stream>>>(x, xbf, (int)NC);
    k_transpose_bf<<<(128 * 128 + 255) / 256, 256, 0, stream>>>(Wg, WgT, 128, 128);
    k_transpose_bf<<<(128 * 128 + 255) / 256, 256, 0, stream>>>(Wh, WhT, 128, 128);
    k_transpose_bf<<<(128 * 128 + 255) / 256, 256, 0, stream>>>(Wo, WoT, 128, 128);
    k_transpose_bf<<<(128 * 512 + 255) / 256, 256, 0, stream>>>(W1, W1T, 128, 512);
    k_transpose_bf<<<(512 * 128 + 255) / 256, 256, 0, stream>>>(W2, W2T, 512, 128);

    // 1) dual projection GEMM (WMMA)
    k_dual_gemm_proj<<<rowBlocks, 256, 0, stream>>>(xbf, WgT, WhT, bg, bh, gproj, hproj);

    // 2) fused depthwise circular 3D conv for both branches
    k_dwconv3d_dual<<<elemBlocks, 256, 0, stream>>>(gproj, hproj, gker, hker, gspat, hspat);

    // 3) temporal log-space scan -> exp(log_h) bf16
    k_scan<<<(Bdim * HWdim * Cdim + 255) / 256, 256, 0, stream>>>(gspat, hspat, expU);

    // 4) Wo GEMM + LayerNorm fused (WMMA)
    k_gemm_wo_ln<<<rowBlocks, 256, 0, stream>>>(expU, WoT, bo, lns, lnb, xn);

    // 5) fused MLP + LayerScale + residual (WMMA, hidden staged in LDS)
    k_mlp<<<rowBlocks, 256, 0, stream>>>(xn, W1T, W2T, b1, b2, gamma, x, out);

    (void)in_sizes; (void)n_in; (void)out_size; (void)ws_size;
}
